// InterpretableAttention_39058432590442
// MI455X (gfx1250) — compile-verified, matrix operand order checked by reading, not run on hardware
//
#include <hip/hip_runtime.h>

// ---------------------------------------------------------------------------
// InterpretableAttention on MI455X (gfx1250), wave32 + WMMA f32<-f16.
//
// Reference math note: the per-row additive bias (feature-MLP -> softmax(faw))
// is constant over the softmax axis j, so softmax(dots + bias) == softmax(dots).
// The whole ext_* / faw path is mathematically dead and is eliminated
// (saves ~161 GFLOP of 425 GFLOP).
//
// Pipeline:
//   1) one-time f32->f16 convert of x; convert+transpose of qkv_w / out_w
//      (so every GEMM operand is row-major [row][K] f16 in global memory)
//   2) fused qkv+attention per (batch, head), WMMA f32<-f16
//   3) output projection GEMM, WMMA f32<-f16
// All LDS tiles are [row][K] f16, so each WMMA fragment is two aligned
// ds_load_b128, and global->LDS staging is pure 16B copies, done with
// GLOBAL_LOAD_ASYNC_TO_LDS_B128 (ASYNCcnt) when the builtin is available.
// ---------------------------------------------------------------------------

typedef __attribute__((ext_vector_type(16))) _Float16 v16h;
typedef __attribute__((ext_vector_type(8)))  _Float16 v8h;
typedef __attribute__((ext_vector_type(4)))  _Float16 v4h;
typedef __attribute__((ext_vector_type(8)))  float    v8f;
typedef __attribute__((ext_vector_type(4)))  int      v4i;

#define DIMD     1024
#define HEADS    16
#define DIM_HEAD 64
#define INNERD   1024
#define SEQ      60
#define BATCH    512
#define ROWS     (BATCH * SEQ)   // 30720

#define AP   72    // LDS pitch in halfs (64 + 8; multiple of 8 -> 16B-aligned rows)
#define QP   72    // pitch for q / k / vT / attn tiles
#define DOTP 65    // dots pitch in floats (odd -> conflict-free column walks)

#if defined(__HIP_DEVICE_COMPILE__) && \
    __has_builtin(__builtin_amdgcn_global_load_async_to_lds_b128) && \
    __has_builtin(__builtin_amdgcn_s_wait_asynccnt)
#define USE_ASYNC 1
#else
#define USE_ASYNC 0
#endif

#define GAS __attribute__((address_space(1)))
#define LAS __attribute__((address_space(3)))

// region_of[token]: region index for channel tokens 7..59 (-1 for region tokens)
__constant__ int c_region_of[SEQ] = {
  -1,-1,-1,-1,-1,-1,-1,
   0, 2, 2, 0, 2, 4, 2, 2, 6, 0, 4, 5, 2, 4, 6, 6, 4, 4, 6, 4,
   6, 6, 6, 5, 4, 5, 6, 6, 6, 6, 4, 4, 6, 4, 6, 6, 6, 5, 4, 1,
   6, 4, 6, 3, 4, 3, 1, 6, 3, 3, 3, 1, 3
};

__device__ inline v8f vzero() {
  v8f z;
#pragma unroll
  for (int i = 0; i < 8; ++i) z[i] = 0.0f;
  return z;
}

__device__ inline v8f wmma_f16(v16h a, v16h b, v8f c) {
  // D = A(16x32 f16) * B(32x16 f16) + C(16x16 f32)
  return __builtin_amdgcn_wmma_f32_16x16x32_f16(
      /*neg_a=*/false, a, /*neg_b=*/false, b,
      /*c_mod=*/(short)0, c, /*reuse_a=*/false, /*reuse_b=*/false);
}

// ---- 16B global->LDS copy: async (ASYNCcnt) when available ----------------
__device__ inline void cp16(_Float16* lds, const _Float16* g) {
#if USE_ASYNC
  __builtin_amdgcn_global_load_async_to_lds_b128(
      (GAS v4i*)g, (LAS v4i*)lds, 0, 0);
#else
  *(v8h*)lds = *(const v8h*)g;
#endif
}
__device__ inline void cp_wait() {
#if USE_ASYNC
  __builtin_amdgcn_s_wait_asynccnt(0);
#endif
}

// ---- WMMA fragment load from LDS [row][K] halfs, pitch P ------------------
// Lane L: row = row0 + L%16, K-half offset = 8*(L/16). The lane's 16 halfs are
// the two contiguous octets at K = k0+h8+{0..7} and K = k0+h8+16+{0..7}
// -> two aligned ds_load_b128.
template <int P>
__device__ inline v16h ldsFrag(const _Float16* base, int row0, int k0) {
  const int lane = threadIdx.x & 31;
  const _Float16* r =
      base + (row0 + (lane & 15)) * P + k0 + ((lane >> 4) << 3);
  const v8h lo = *(const v8h*)(r);
  const v8h hi = *(const v8h*)(r + 16);
  return __builtin_shufflevector(lo, hi, 0, 1, 2, 3, 4, 5, 6, 7,
                                         8, 9, 10, 11, 12, 13, 14, 15);
}

// ---- stage a 64x64 f16 tile: global [row][ld] -> LDS [row][K], pitch P ----
// 256 threads x 2 iterations x 16B, all addresses 16B-aligned.
template <int P>
__device__ inline void stage16(_Float16* lds, const _Float16* __restrict__ g,
                               long long row0, int k0, int ld) {
  const int t = threadIdx.x;
#pragma unroll
  for (int it = 0; it < 2; ++it) {
    const int o = t + it * 256;
    const int r = o >> 3;
    const int c = (o & 7) << 3;
    cp16(lds + r * P + c, g + (row0 + r) * (long long)ld + k0 + c);
  }
}

// ---------------------------------------------------------------------------
// Precision/layout prep kernels (one-time, bandwidth-trivial)
// ---------------------------------------------------------------------------
__global__ __launch_bounds__(256) void cvt_f32_f16_kernel(
    const float* __restrict__ in, _Float16* __restrict__ out, long long nquads) {
  const long long i = (long long)blockIdx.x * 256 + threadIdx.x;
  if (i >= nquads) return;
  const float4 x = ((const float4*)in)[i];
  v4h p;
  p[0] = (_Float16)x.x; p[1] = (_Float16)x.y;
  p[2] = (_Float16)x.z; p[3] = (_Float16)x.w;
  ((v4h*)out)[i] = p;
}

// in[K][N] f32 -> out[N][K] f16 (64x64 LDS tile transpose)
__global__ __launch_bounds__(256) void cvtT_f32_f16_kernel(
    const float* __restrict__ in, _Float16* __restrict__ out, int K, int N) {
  __shared__ _Float16 tile[64 * AP];  // [n][k]
  const int t = threadIdx.x;
  const int n0 = blockIdx.x * 64;
  const int k0 = blockIdx.y * 64;
#pragma unroll
  for (int it = 0; it < 4; ++it) {
    const int r = (t >> 4) + it * 16;     // k
    const int c = (t & 15) << 2;          // n
    const float4 x = *(const float4*)(in + (long long)(k0 + r) * N + n0 + c);
    tile[(c + 0) * AP + r] = (_Float16)x.x;
    tile[(c + 1) * AP + r] = (_Float16)x.y;
    tile[(c + 2) * AP + r] = (_Float16)x.z;
    tile[(c + 3) * AP + r] = (_Float16)x.w;
  }
  __syncthreads();
#pragma unroll
  for (int it = 0; it < 2; ++it) {
    const int o = t + it * 256;
    const int n = o >> 3;
    const int c8 = (o & 7) << 3;
    *(v8h*)(out + (long long)(n0 + n) * K + k0 + c8) =
        *(const v8h*)(tile + n * AP + c8);
  }
}

// ---------------------------------------------------------------------------
// Kernel 1: per-(batch, head) fused qkv projection + masked attention.
// Writes inner[b*60 + i, h*64 + d] (f16) to workspace.
// Pad rows 60..63 read adjacent workspace bytes; their products are always
// discarded (rows dropped, cols masked, attn cols zeroed), so no guards.
// ---------------------------------------------------------------------------
struct SMem {  // reuse staging LDS for softmax buffers
  union {
    struct { _Float16 A[64 * AP]; _Float16 BT[192 * AP]; } st;  // 36864 B
    struct { float d[64 * DOTP];  _Float16 at[64 * QP];  } sm;  // 25856 B
  };
};

__global__ __launch_bounds__(256) void attn_fused_kernel(
    const _Float16* __restrict__ xh, const _Float16* __restrict__ qkvT,
    _Float16* __restrict__ inner) {
  const int h = blockIdx.x;
  const int b = blockIdx.y;
  const int tid = threadIdx.x;
  const int w = tid >> 5;
  const int lane = tid & 31;
  const int lhalf = lane >> 4;
  const int ln = lane & 15;

  __shared__ _Float16 qlds[64 * QP];   // q[i][d]  (A layout for q@kT), *scale
  __shared__ _Float16 klds[64 * QP];   // k[j][d]  (BT layout for q@kT)
  __shared__ _Float16 vtlds[64 * QP];  // vT[d][j] (BT layout for attn@v)
  __shared__ unsigned long long smask[64];
  __shared__ SMem u;

  // Build structured region-channel interaction mask (one 60-bit row mask each)
  if (tid < SEQ) {
    unsigned long long m = 0;
    for (int j = 0; j < SEQ; ++j) {
      const bool ir = tid < 7, jr = j < 7;
      const bool on = (ir == jr) || (ir ? (c_region_of[j] == tid)
                                        : (c_region_of[tid] == j));
      if (on) m |= 1ull << j;
    }
    smask[tid] = m;
  }

  const _Float16* xb = xh + (long long)b * SEQ * DIMD;

  // -------- qkv: [64(60) x 192] = x_b[64 x 1024] @ qkv_w[:, head slices] ----
  v8f acc[6];
#pragma unroll
  for (int t = 0; t < 6; ++t) acc[t] = vzero();
  const int mT = w & 3;              // wave's 16-row tile
  const int gN = (w >> 2) * 6;       // wave's 6 column tiles (of 12)

  for (int k0 = 0; k0 < DIMD; k0 += 64) {
    stage16<AP>(u.st.A, xb, 0, k0, DIMD);
#pragma unroll
    for (int g = 0; g < 3; ++g)  // q | k | v row groups of qkvT
      stage16<AP>(u.st.BT + g * 64 * AP, qkvT,
                  (long long)g * INNERD + h * DIM_HEAD, k0, DIMD);
    cp_wait();
    __syncthreads();
    const v16h a0 = ldsFrag<AP>(u.st.A, mT * 16, 0);
    const v16h a1 = ldsFrag<AP>(u.st.A, mT * 16, 32);
#pragma unroll
    for (int t = 0; t < 6; ++t) {
      const int n0 = (gN + t) * 16;
      acc[t] = wmma_f16(a0, ldsFrag<AP>(u.st.BT, n0, 0), acc[t]);
      acc[t] = wmma_f16(a1, ldsFrag<AP>(u.st.BT, n0, 32), acc[t]);
    }
    __syncthreads();
  }

  // Scatter q (pre-scaled by dim_head^-0.5), k, vT into f16 LDS tiles
#pragma unroll
  for (int t = 0; t < 6; ++t) {
    const int N = (gN + t) * 16 + ln;
#pragma unroll
    for (int r = 0; r < 8; ++r) {
      const int M = mT * 16 + r + 8 * lhalf;
      const float vv = acc[t][r];
      if (N < 64)        qlds[M * QP + N]          = (_Float16)(vv * 0.125f);
      else if (N < 128)  klds[M * QP + (N - 64)]   = (_Float16)vv;   // k[j][d]
      else               vtlds[(N - 128) * QP + M] = (_Float16)vv;   // vT[d][j]
    }
  }
  __syncthreads();

  // -------- dots = q @ k^T, masked ------------------------------------------
  v8f d0 = vzero(), d1 = vzero();
  {
    const v16h a0 = ldsFrag<QP>(qlds, mT * 16, 0);
    const v16h a1 = ldsFrag<QP>(qlds, mT * 16, 32);
    const int n0 = ((w >> 2) * 2) * 16, n1 = n0 + 16;
    d0 = wmma_f16(a0, ldsFrag<QP>(klds, n0, 0), d0);
    d0 = wmma_f16(a1, ldsFrag<QP>(klds, n0, 32), d0);
    d1 = wmma_f16(a0, ldsFrag<QP>(klds, n1, 0), d1);
    d1 = wmma_f16(a1, ldsFrag<QP>(klds, n1, 32), d1);
  }
#pragma unroll
  for (int t = 0; t < 2; ++t) {
    const v8f dd = t ? d1 : d0;
    const int N = ((w >> 2) * 2 + t) * 16 + ln;
#pragma unroll
    for (int r = 0; r < 8; ++r) {
      const int M = mT * 16 + r + 8 * lhalf;
      const bool on = (M < SEQ) && (N < SEQ) && ((smask[M] >> N) & 1ull);
      u.sm.d[M * DOTP + N] = on ? dd[r] : -3.0e38f;
    }
  }
  __syncthreads();

  // -------- row softmax (row-constant bias is softmax-invariant: dropped) ---
  if (tid < 64) {
    const int i = tid;
    if (i < SEQ) {
      float mx = -3.4e38f;
      for (int j = 0; j < SEQ; ++j) mx = fmaxf(mx, u.sm.d[i * DOTP + j]);
      float s = 0.f;
      for (int j = 0; j < SEQ; ++j) {
        const float e = expf(u.sm.d[i * DOTP + j] - mx);
        u.sm.d[i * DOTP + j] = e;
        s += e;
      }
      const float inv = 1.f / s;
      for (int j = 0; j < SEQ; ++j)
        u.sm.at[i * QP + j] = (_Float16)(u.sm.d[i * DOTP + j] * inv);
      for (int j = SEQ; j < 64; ++j) u.sm.at[i * QP + j] = (_Float16)0.f;
    } else {
      for (int j = 0; j < 64; ++j) u.sm.at[i * QP + j] = (_Float16)0.f;
    }
  }
  __syncthreads();

  // -------- out_head = attn @ v --------------------------------------------
  v8f o0 = vzero(), o1 = vzero();
  {
    const v16h a0 = ldsFrag<QP>(u.sm.at, mT * 16, 0);
    const v16h a1 = ldsFrag<QP>(u.sm.at, mT * 16, 32);
    const int n0 = ((w >> 2) * 2) * 16, n1 = n0 + 16;
    o0 = wmma_f16(a0, ldsFrag<QP>(vtlds, n0, 0), o0);
    o0 = wmma_f16(a1, ldsFrag<QP>(vtlds, n0, 32), o0);
    o1 = wmma_f16(a0, ldsFrag<QP>(vtlds, n1, 0), o1);
    o1 = wmma_f16(a1, ldsFrag<QP>(vtlds, n1, 32), o1);
  }
  _Float16* ib = inner + ((long long)b * SEQ) * INNERD + h * DIM_HEAD;
#pragma unroll
  for (int t = 0; t < 2; ++t) {
    const v8f oo = t ? o1 : o0;
    const int N = ((w >> 2) * 2 + t) * 16 + ln;
#pragma unroll
    for (int r = 0; r < 8; ++r) {
      const int M = mT * 16 + r + 8 * lhalf;
      if (M < SEQ) ib[(long long)M * INNERD + N] = (_Float16)oo[r];
    }
  }
}

// ---------------------------------------------------------------------------
// Kernel 2: out = inner[30720,1024](f16) @ owT^T + out_b   (f32 out)
// ---------------------------------------------------------------------------
__global__ __launch_bounds__(256) void out_proj_kernel(
    const _Float16* __restrict__ inner, const _Float16* __restrict__ owT,
    const float* __restrict__ ob, float* __restrict__ out) {
  const int tid = threadIdx.x;
  const int w = tid >> 5;
  const int lane = tid & 31;
  const int lhalf = lane >> 4;
  const int ln = lane & 15;
  const long long row0 = (long long)blockIdx.y * 64;
  const int col0 = blockIdx.x * 64;

  __shared__ _Float16 As[64 * AP];
  __shared__ _Float16 BTs[64 * AP];

  v8f c0 = vzero(), c1 = vzero();
  const int mT = w & 3;
  const int nb = (w >> 2) * 2;

  for (int k0 = 0; k0 < INNERD; k0 += 64) {
    stage16<AP>(As, inner, row0, k0, INNERD);
    stage16<AP>(BTs, owT, col0, k0, DIMD);
    cp_wait();
    __syncthreads();
    const v16h a0 = ldsFrag<AP>(As, mT * 16, 0);
    const v16h a1 = ldsFrag<AP>(As, mT * 16, 32);
    c0 = wmma_f16(a0, ldsFrag<AP>(BTs, nb * 16, 0), c0);
    c0 = wmma_f16(a1, ldsFrag<AP>(BTs, nb * 16, 32), c0);
    c1 = wmma_f16(a0, ldsFrag<AP>(BTs, (nb + 1) * 16, 0), c1);
    c1 = wmma_f16(a1, ldsFrag<AP>(BTs, (nb + 1) * 16, 32), c1);
    __syncthreads();
  }

#pragma unroll
  for (int t = 0; t < 2; ++t) {
    const v8f cc = t ? c1 : c0;
    const int col = col0 + (nb + t) * 16 + ln;
#pragma unroll
    for (int r = 0; r < 8; ++r) {
      const int M = mT * 16 + r + 8 * lhalf;
      out[(row0 + M) * (long long)DIMD + col] = cc[r] + ob[col];
    }
  }
}

// ---------------------------------------------------------------------------
extern "C" void kernel_launch(void* const* d_in, const int* in_sizes, int n_in,
                              void* d_out, int out_size, void* d_ws, size_t ws_size,
                              hipStream_t stream) {
  (void)in_sizes; (void)n_in; (void)out_size; (void)ws_size;
  const float* x     = (const float*)d_in[0];
  // d_in[1..4] = ext_w1/ext_b1/ext_w2/ext_b2: mathematically dead (row-constant
  // bias inside a softmax over the last axis). d_in[8] = faw: likewise dead.
  const float* qkv_w = (const float*)d_in[5];
  const float* out_w = (const float*)d_in[6];
  const float* out_b = (const float*)d_in[7];

  // workspace layout (f16, all 16B-aligned): ~134 MB total
  _Float16* xh    = (_Float16*)d_ws;                       // [30720][1024]
  _Float16* qkvT  = xh   + (long long)ROWS * DIMD;         // [3072][1024]
  _Float16* owT   = qkvT + (long long)3 * INNERD * DIMD;   // [1024][1024]
  _Float16* innerh= owT  + (long long)DIMD * INNERD;       // [30720][1024]

  const long long xquads = (long long)ROWS * DIMD / 4;
  cvt_f32_f16_kernel<<<(unsigned)((xquads + 255) / 256), 256, 0, stream>>>(
      x, xh, xquads);
  cvtT_f32_f16_kernel<<<dim3(3 * INNERD / 64, DIMD / 64), 256, 0, stream>>>(
      qkv_w, qkvT, DIMD, 3 * INNERD);
  cvtT_f32_f16_kernel<<<dim3(INNERD / 64, DIMD / 64), 256, 0, stream>>>(
      out_w, owT, DIMD, INNERD);

  attn_fused_kernel<<<dim3(HEADS, BATCH), 256, 0, stream>>>(xh, qkvT, innerh);
  out_proj_kernel<<<dim3(INNERD / 64, ROWS / 64), 256, 0, stream>>>(
      innerh, owT, out_b, (float*)d_out);
}
